// Net_66073776882394
// MI455X (gfx1250) — matrix operand order repaired and checked by reading, not compiled, not run on hardware
//
#include <hip/hip_runtime.h>
#include <hip/hip_bf16.h>

typedef _Float16 v16h __attribute__((ext_vector_type(16)));
typedef float    v8f  __attribute__((ext_vector_type(8)));

#define NNODES 100000
#define NEDGES 1600000
#define ETOT   (NEDGES + NNODES)
#define FIN    500
#define HEADS  8
#define NEG_SLOPE 0.2f
#define ENC_NEG_INF 0x007FFFFFu   // encf(-inf)

// ---------- order-preserving float <-> uint for atomicMax on signed floats ----
static __device__ __forceinline__ unsigned encf(float f) {
  unsigned u = __float_as_uint(f);
  return (u & 0x80000000u) ? ~u : (u | 0x80000000u);
}
static __device__ __forceinline__ float decf(unsigned u) {
  u = (u & 0x80000000u) ? (u & 0x7FFFFFFFu) : ~u;
  return __uint_as_float(u);
}

// ---------------------------- fills ------------------------------------------
__global__ void fill_u32_kernel(unsigned* __restrict__ p, unsigned v, int n) {
  int i = blockIdx.x * blockDim.x + threadIdx.x;
  if (i < n) p[i] = v;
}

// ------------- W -> transposed, K-padded f16  (Wt[col][KPAD]) ----------------
__global__ void convert_wt_kernel(const float* __restrict__ W,
                                  _Float16* __restrict__ Wt,
                                  int K, int ncolValid, int KPAD, int total) {
  int t = blockIdx.x * blockDim.x + threadIdx.x;
  if (t >= total) return;
  int c = t / KPAD, k = t - c * KPAD;
  float v = (k < K && c < ncolValid) ? W[k * ncolValid + c] : 0.0f;
  Wt[t] = (_Float16)v;
}

// ---------------------------- WMMA GEMM (direct) -----------------------------
// Y[16*grid, ncolValid] = X[., KTOT](f32) * Wt^T  with Wt f16 [ncolPad][KPAD].
// No LDS: A and B both use contiguous k order (k = 16h + s), so each lane's fragments are single runs; we
// load them directly (b128), convert A f32->f16 in regs, and issue v_wmma.
// blockDim.x = 32 * numColTiles; wave w owns output cols [16w,16w+16).
template <int KTOT, int KPAD>
__global__ void gemm_wmma_direct(const float* __restrict__ X, int ldx,
                                 const _Float16* __restrict__ Wt,
                                 float* __restrict__ Y, int ldy,
                                 int ncolValid) {
  const int lane    = threadIdx.x & 31;
  const int wave    = threadIdx.x >> 5;
  const int rowBase = blockIdx.x * 16;
  const int arow    = rowBase + (lane & 15);
  const int akb     = (lane >= 16) ? 16 : 0;   // A: K sub-block per lane half
  const int bcol    = (wave << 4) + (lane & 15);
  const int bkb     = (lane >= 16) ? 16 : 0;   // B: K sub-block per lane half

  const float*    xr = X + (size_t)arow * ldx;
  const _Float16* wc = Wt + (size_t)bcol * KPAD;

  v8f acc = {};
  constexpr int NFULL = KTOT / 32;

  for (int ch = 0; ch < NFULL; ++ch) {
    const int k0 = ch * 32;
    // A fragment: one contiguous 16-f32 run -> 4x b128 loads, cvt to f16
    float4 a0 = *(const float4*)(xr + k0 + akb);
    float4 a1 = *(const float4*)(xr + k0 + akb + 4);
    float4 a2 = *(const float4*)(xr + k0 + akb + 8);
    float4 a3 = *(const float4*)(xr + k0 + akb + 12);
    v16h afrag;
    afrag[0]  = (_Float16)a0.x; afrag[1]  = (_Float16)a0.y;
    afrag[2]  = (_Float16)a0.z; afrag[3]  = (_Float16)a0.w;
    afrag[4]  = (_Float16)a1.x; afrag[5]  = (_Float16)a1.y;
    afrag[6]  = (_Float16)a1.z; afrag[7]  = (_Float16)a1.w;
    afrag[8]  = (_Float16)a2.x; afrag[9]  = (_Float16)a2.y;
    afrag[10] = (_Float16)a2.z; afrag[11] = (_Float16)a2.w;
    afrag[12] = (_Float16)a3.x; afrag[13] = (_Float16)a3.y;
    afrag[14] = (_Float16)a3.z; afrag[15] = (_Float16)a3.w;
    // B fragment: one contiguous 16-f16 run (32B aligned) -> 2x b128 loads
    v16h bfrag = *(const v16h*)(wc + k0 + bkb);
    acc = __builtin_amdgcn_wmma_f32_16x16x32_f16(false, afrag, false, bfrag,
                                                 (short)0, acc, false, false);
  }

  if constexpr (KTOT % 32 != 0) {   // tail chunk: clamp+select, no branches
    const int k0 = NFULL * 32;
    v16h afrag;
#pragma unroll
    for (int e = 0; e < 16; ++e) {
      int k  = k0 + akb + e;
      int kc = (k < KTOT) ? k : (KTOT - 1);
      float xv = xr[kc];
      afrag[e] = (_Float16)((k < KTOT) ? xv : 0.0f);
    }
    v16h bfrag = *(const v16h*)(wc + k0 + bkb);  // Wt zero-padded to KPAD
    acc = __builtin_amdgcn_wmma_f32_16x16x32_f16(false, afrag, false, bfrag,
                                                 (short)0, acc, false, false);
  }

  // C/D layout: VGPR v -> row v (+8 for lanes 16-31), col = lane%16
  const int col = (wave << 4) + (lane & 15);
  if (col < ncolValid) {
    const int rOff = (lane >= 16) ? 8 : 0;
#pragma unroll
    for (int vg = 0; vg < 8; ++vg)
      Y[(size_t)(rowBase + vg + rOff) * ldy + col] = acc[vg];
  }
}

// --------------------- per-node attention scores -----------------------------
__global__ void score_kernel(const float* __restrict__ xp,
                             const float* __restrict__ asrc,
                             const float* __restrict__ adst,
                             float* __restrict__ es, float* __restrict__ ed,
                             int O, int total) {
  int t = blockIdx.x * blockDim.x + threadIdx.x;
  if (t >= total) return;
  int n = t >> 3, h = t & 7;
  const float* xb = xp + (n * HEADS + h) * O;
  float s = 0.f, d = 0.f;
  for (int o = 0; o < O; ++o) {
    float xv = xb[o];
    s += xv * asrc[h * O + o];
    d += xv * adst[h * O + o];
  }
  es[t] = s;
  ed[t] = d;
}

// ------------------------------ edge passes ----------------------------------
static __device__ __forceinline__ void edge_ids(const int* __restrict__ ei,
                                                int e, int& src, int& dst) {
  if (e < NEDGES) { src = ei[e]; dst = ei[NEDGES + e]; }
  else            { src = e - NEDGES; dst = src; }        // self loops
}
static __device__ __forceinline__ float edge_score(const float* es,
                                                   const float* ed,
                                                   int src, int dst, int h) {
  float s = es[src * HEADS + h] + ed[dst * HEADS + h];
  return (s > 0.f) ? s : NEG_SLOPE * s;                   // leaky relu
}

__global__ void edge_max_kernel(const int* __restrict__ ei,
                                const float* __restrict__ es,
                                const float* __restrict__ ed,
                                unsigned* __restrict__ m) {
  int t = blockIdx.x * blockDim.x + threadIdx.x;
  if (t >= ETOT * HEADS) return;
  int e = t >> 3, h = t & 7, src, dst;
  edge_ids(ei, e, src, dst);
  atomicMax(&m[dst * HEADS + h], encf(edge_score(es, ed, src, dst, h)));
}

__global__ void edge_sum_kernel(const int* __restrict__ ei,
                                const float* __restrict__ es,
                                const float* __restrict__ ed,
                                const unsigned* __restrict__ m,
                                float* __restrict__ den) {
  int t = blockIdx.x * blockDim.x + threadIdx.x;
  if (t >= ETOT * HEADS) return;
  int e = t >> 3, h = t & 7, src, dst;
  edge_ids(ei, e, src, dst);
  float s = edge_score(es, ed, src, dst, h);
  atomicAdd(&den[dst * HEADS + h], __expf(s - decf(m[dst * HEADS + h])));
}

template <int O>
__global__ void edge_agg_kernel(const int* __restrict__ ei,
                                const float* __restrict__ es,
                                const float* __restrict__ ed,
                                const unsigned* __restrict__ m,
                                const float* __restrict__ den,
                                const float* __restrict__ xp,
                                float* __restrict__ agg) {
  int t = blockIdx.x * blockDim.x + threadIdx.x;
  if (t >= ETOT * HEADS) return;
  int e = t >> 3, h = t & 7, src, dst;
  edge_ids(ei, e, src, dst);
  float s     = edge_score(es, ed, src, dst, h);
  float alpha = __expf(s - decf(m[dst * HEADS + h])) / den[dst * HEADS + h];
  const float* xb = xp  + (src * HEADS + h) * O;
  float*       ab = agg + (dst * HEADS + h) * O;
  if constexpr (O == 8) {           // 32B-aligned: two float4 gathers
    float4 v0 = *(const float4*)(xb);
    float4 v1 = *(const float4*)(xb + 4);
    atomicAdd(&ab[0], v0.x * alpha); atomicAdd(&ab[1], v0.y * alpha);
    atomicAdd(&ab[2], v0.z * alpha); atomicAdd(&ab[3], v0.w * alpha);
    atomicAdd(&ab[4], v1.x * alpha); atomicAdd(&ab[5], v1.y * alpha);
    atomicAdd(&ab[6], v1.z * alpha); atomicAdd(&ab[7], v1.w * alpha);
  } else {
#pragma unroll
    for (int o = 0; o < O; ++o) atomicAdd(&ab[o], xb[o] * alpha);
  }
}

// --------------------------- bias + ELU (layer 1) ----------------------------
__global__ void bias_elu_kernel(float* __restrict__ agg,
                                const float* __restrict__ b, int total) {
  int t = blockIdx.x * blockDim.x + threadIdx.x;
  if (t >= total) return;
  float v = agg[t] + b[t & 63];                  // 64 = H*O1 features
  agg[t] = (v > 0.f) ? v : (__expf(v) - 1.f);
}

// --------------- head mean + bias + log_softmax (layer 2 out) ----------------
__global__ void final_kernel(const float* __restrict__ agg2,
                             const float* __restrict__ b2,
                             float* __restrict__ out) {
  int n = blockIdx.x * blockDim.x + threadIdx.x;
  if (n >= NNODES) return;
  float v[3];
#pragma unroll
  for (int c = 0; c < 3; ++c) {
    float s = 0.f;
#pragma unroll
    for (int h = 0; h < HEADS; ++h) s += agg2[(n * HEADS + h) * 3 + c];
    v[c] = s * (1.f / 8.f) + b2[c];
  }
  float mx  = fmaxf(v[0], fmaxf(v[1], v[2]));
  float lse = __logf(__expf(v[0] - mx) + __expf(v[1] - mx) + __expf(v[2] - mx));
#pragma unroll
  for (int c = 0; c < 3; ++c) out[n * 3 + c] = v[c] - mx - lse;
}

// ----------------------------- launcher --------------------------------------
extern "C" void kernel_launch(void* const* d_in, const int* in_sizes, int n_in,
                              void* d_out, int out_size, void* d_ws, size_t ws_size,
                              hipStream_t stream) {
  (void)in_sizes; (void)n_in; (void)out_size; (void)ws_size;
  const float* x   = (const float*)d_in[0];
  const int*   ei  = (const int*)  d_in[1];
  const float* W1  = (const float*)d_in[2];
  const float* as1 = (const float*)d_in[3];
  const float* ad1 = (const float*)d_in[4];
  const float* b1  = (const float*)d_in[5];
  const float* W2  = (const float*)d_in[6];
  const float* as2 = (const float*)d_in[7];
  const float* ad2 = (const float*)d_in[8];
  const float* b2  = (const float*)d_in[9];
  float* out = (float*)d_out;

  char* ws = (char*)d_ws;
  size_t off = 0;
  auto carve = [&](size_t bytes) {
    char* p = ws + off;
    off = (off + bytes + 255) & ~(size_t)255;
    return p;
  };
  float*     xp1  = (float*)    carve((size_t)NNODES * 64 * 4);
  float*     es1  = (float*)    carve((size_t)NNODES * 8 * 4);
  float*     ed1  = (float*)    carve((size_t)NNODES * 8 * 4);
  unsigned*  m1   = (unsigned*) carve((size_t)NNODES * 8 * 4);
  float*     den1 = (float*)    carve((size_t)NNODES * 8 * 4);
  float*     agg1 = (float*)    carve((size_t)NNODES * 64 * 4); // -> h after ELU
  float*     xp2  = (float*)    carve((size_t)NNODES * 24 * 4);
  float*     es2  = (float*)    carve((size_t)NNODES * 8 * 4);
  float*     ed2  = (float*)    carve((size_t)NNODES * 8 * 4);
  unsigned*  m2   = (unsigned*) carve((size_t)NNODES * 8 * 4);
  float*     den2 = (float*)    carve((size_t)NNODES * 8 * 4);
  float*     agg2 = (float*)    carve((size_t)NNODES * 24 * 4);
  _Float16*  Wt1  = (_Float16*) carve((size_t)64 * 512 * 2);    // [64][512]
  _Float16*  Wt2  = (_Float16*) carve((size_t)32 * 64 * 2);     // [32][64]

  const int TB = 256;
  const int nNH  = NNODES * 8;
  const int nN64 = NNODES * 64;
  const int nN24 = NNODES * 24;
  const int nEH  = ETOT * 8;
  const int gNH  = (nNH  + TB - 1) / TB;
  const int gN64 = (nN64 + TB - 1) / TB;
  const int gN24 = (nN24 + TB - 1) / TB;
  const int gEH  = (nEH  + TB - 1) / TB;
  const int gN   = (NNODES + TB - 1) / TB;
  const int gemmBlocks = NNODES / 16;     // 100000/16 = 6250, exact

  // init accumulators (d_ws is poisoned by the harness)
  fill_u32_kernel<<<gNH,  TB, 0, stream>>>(m1, ENC_NEG_INF, nNH);
  fill_u32_kernel<<<gNH,  TB, 0, stream>>>((unsigned*)den1, 0u, nNH);
  fill_u32_kernel<<<gN64, TB, 0, stream>>>((unsigned*)agg1, 0u, nN64);
  fill_u32_kernel<<<gNH,  TB, 0, stream>>>(m2, ENC_NEG_INF, nNH);
  fill_u32_kernel<<<gNH,  TB, 0, stream>>>((unsigned*)den2, 0u, nNH);
  fill_u32_kernel<<<gN24, TB, 0, stream>>>((unsigned*)agg2, 0u, nN24);

  // transpose + f16-pad weights (tiny, stays in L2)
  convert_wt_kernel<<<(64 * 512 + TB - 1) / TB, TB, 0, stream>>>(
      W1, Wt1, FIN, 64, 512, 64 * 512);
  convert_wt_kernel<<<(32 * 64 + TB - 1) / TB, TB, 0, stream>>>(
      W2, Wt2, 64, 24, 64, 32 * 64);

  // ---- layer 1 ----
  gemm_wmma_direct<FIN, 512><<<gemmBlocks, 128, 0, stream>>>(x, FIN, Wt1,
                                                             xp1, 64, 64);
  score_kernel<<<gNH, TB, 0, stream>>>(xp1, as1, ad1, es1, ed1, 8, nNH);
  edge_max_kernel<<<gEH, TB, 0, stream>>>(ei, es1, ed1, m1);
  edge_sum_kernel<<<gEH, TB, 0, stream>>>(ei, es1, ed1, m1, den1);
  edge_agg_kernel<8><<<gEH, TB, 0, stream>>>(ei, es1, ed1, m1, den1, xp1, agg1);
  bias_elu_kernel<<<gN64, TB, 0, stream>>>(agg1, b1, nN64);

  // ---- layer 2 ----
  gemm_wmma_direct<64, 64><<<gemmBlocks, 64, 0, stream>>>(agg1, 64, Wt2,
                                                          xp2, 24, 24);
  score_kernel<<<gNH, TB, 0, stream>>>(xp2, as2, ad2, es2, ed2, 3, nNH);
  edge_max_kernel<<<gEH, TB, 0, stream>>>(ei, es2, ed2, m2);
  edge_sum_kernel<<<gEH, TB, 0, stream>>>(ei, es2, ed2, m2, den2);
  edge_agg_kernel<3><<<gEH, TB, 0, stream>>>(ei, es2, ed2, m2, den2, xp2, agg2);

  final_kernel<<<gN, TB, 0, stream>>>(agg2, b2, out);
}
